// LDSTNet_35046933135565
// MI455X (gfx1250) — compile-verified, matrix operand order checked by reading, not run on hardware
//
#include <hip/hip_runtime.h>
#include <hip/hip_bf16.h>

// MI455X (gfx1250) implementation. All heavy contractions use
// v_wmma_f32_16x16x32_f16 (f16 in, f32 accumulate). wave32 everywhere.
// Fragment gathers are branchless; full 32-wide K blocks skip all masking
// (out-of-range rows/cols only feed D elements that are never stored), so
// the steady-state inner loop is load -> cvt_f16 -> v_wmma only.

typedef __attribute__((ext_vector_type(16))) _Float16 v16h;
typedef __attribute__((ext_vector_type(8)))  float    v8f;

#define BN_SCALE 0.9999950000374997f

struct GemmDesc {
  const float* A; const float* B; float* C;
  int M, N, K;
  long long sAb0, sAb1; int sAm, sAk;   // A[b0,b1,m,k]
  long long sBb0, sBb1; int sBk, sBn;   // B[b0,b1,k,n]
  long long sCb0, sCb1; int sCm, sCn;   // C[b0,b1,m,n]
  float alpha;
  const float* biasN;     // optional per-n bias
  const float* preluA;    // epilogue 1 slope (device scalar)
  int epilogue;           // 0 none, 1 prelu(*preluA), 2 relu(v*BN_SCALE)
};

// One wave computes one 16x16 tile of C = alpha*A*B (+bias, epilogue).
// A fragment: lane (hi=lane>>4, lid=lane&15) holds row m=tm+lid, elements j:
//   k = 16*(j>>3) + 8*hi + (j&7)   (ISA 7.12.2, 16-bit A 16x32 layout)
// B fragment mirrored (lane = column n). C/D: acc[r] -> m = r + 8*hi, n = lid.
__global__ __launch_bounds__(32) void gemm_wmma_kernel(GemmDesc d) {
  const int tilesN = (d.N + 15) >> 4;
  const int tile = blockIdx.x;
  const int tm = (tile / tilesN) << 4;
  const int tn = (tile % tilesN) << 4;
  const float* A = d.A + (long long)blockIdx.z * d.sAb0 + (long long)blockIdx.y * d.sAb1;
  const float* B = d.B + (long long)blockIdx.z * d.sBb0 + (long long)blockIdx.y * d.sBb1;
  float*       C = d.C + (long long)blockIdx.z * d.sCb0 + (long long)blockIdx.y * d.sCb1;
  const int lane = threadIdx.x;
  const int half = lane >> 4;
  const int lid  = lane & 15;
  const int am = tm + lid;
  const int bn = tn + lid;
  const bool aok = am < d.M;
  const bool bok = bn < d.N;
  // Clamped row/col base pointers keep every load in-bounds; values from
  // clamped rows/cols only reach D entries that are never stored.
  const float* Arow = A + (aok ? am : 0) * d.sAm;
  const float* Bcol = B + (bok ? bn : 0) * d.sBn;
  const int kl = (half << 3);   // per-lane k offset within the 16-wide group
  v8f acc = {};
  for (int k0 = 0; k0 < d.K; k0 += 32) {
    v16h af, bf;
    if (k0 + 32 <= d.K) {       // uniform: full block, no masking at all
#pragma unroll
      for (int j = 0; j < 16; ++j) {
        const int kk = k0 + ((j >> 3) << 4) + kl + (j & 7);
        af[j] = (_Float16)Arow[kk * d.sAk];
        bf[j] = (_Float16)Bcol[kk * d.sBk];
      }
    } else {                    // k tail: zero both operands past K
#pragma unroll
      for (int j = 0; j < 16; ++j) {
        const int kk = k0 + ((j >> 3) << 4) + kl + (j & 7);
        const bool kok = kk < d.K;
        const int kc = kok ? kk : 0;
        float a = Arow[kc * d.sAk];
        float b = Bcol[kc * d.sBk];
        af[j] = (_Float16)(kok ? a : 0.f);
        bf[j] = (_Float16)(kok ? b : 0.f);
      }
    }
    acc = __builtin_amdgcn_wmma_f32_16x16x32_f16(false, af, false, bf,
                                                 (short)0, acc, false, false);
  }
  if (bok) {
    const float bias = d.biasN ? d.biasN[bn] : 0.f;
    const float pa = (d.epilogue == 1) ? *d.preluA : 0.f;
#pragma unroll
    for (int r = 0; r < 8; ++r) {
      const int cm = tm + r + (half << 3);
      if (cm < d.M) {
        float v = acc[r] * d.alpha + bias;
        if (d.epilogue == 1)      v = v >= 0.f ? v : pa * v;
        else if (d.epilogue == 2) { v *= BN_SCALE; v = v > 0.f ? v : 0.f; }
        C[cm * d.sCm + bn * d.sCn] = v;
      }
    }
  }
}

// Row softmax over last dim L (<=25), optionally += addM[(row%addRows)*L + j].
__global__ __launch_bounds__(32) void softmax_rows(float* data, int L,
                                                   const float* addM, int addRows) {
  long long row = blockIdx.x;
  int lane = threadIdx.x;
  float* p = data + row * L;
  float v = (lane < L) ? p[lane < L ? lane : 0] : -3.4e38f;
  float m = v;
  for (int o = 16; o > 0; o >>= 1) m = fmaxf(m, __shfl_xor(m, o));
  float e = (lane < L) ? __expf(v - m) : 0.f;
  float s = e;
  for (int o = 16; o > 0; o >>= 1) s += __shfl_xor(s, o);
  if (lane < L) {
    float r = e / s;
    if (addM) r += addM[(row % addRows) * L + lane];
    p[lane] = r;
  }
}

// x (B,C,T,N) -> Hcat slice 0 (b-stride 409600)
__global__ void copy_x_hcat(const float* __restrict__ x, float* __restrict__ hcat) {
  long long i = (long long)blockIdx.x * blockDim.x + threadIdx.x;
  if (i >= 6553600LL) return;
  long long b = i / 102400, r = i - b * 102400;
  hcat[b * 409600 + r] = x[i];
}

// Raw-reshape permute for JointForcePooling:
// G[b, i*25+j] with (c,n,t) flattened as c*1600 + n*64 + t.
__global__ void permute_pool(const float* __restrict__ Fp, float* __restrict__ G) {
  long long i = (long long)blockIdx.x * blockDim.x + threadIdx.x;
  if (i >= 6553600LL) return;
  long long b = i / 102400, f = i - b * 102400;
  long long c = f / 1600, rem = f - c * 1600;
  long long n = rem >> 6, t = rem & 63;
  G[i] = Fp[b * 102400 + c * 1600 + t * 25 + n];
}

// Collapse wcat . (Wt x + bt) into a single reduced weight vector.
// out[0..63]=wtv, [64..127]=wpv, [128]=wcat[:Ci].bt, [129]=wcat[Ci:].bph
__global__ void wtv_kernel(const float* wcat, const float* Wt, const float* bt,
                           const float* Wph, const float* bph, float* out) {
  int c = threadIdx.x;  // 64 threads
  float s1 = 0.f, s2 = 0.f;
  for (int i = 0; i < 32; ++i) {
    s1 += wcat[i]      * Wt[i * 64 + c];
    s2 += wcat[32 + i] * Wph[i * 64 + c];
  }
  out[c] = s1;
  out[64 + c] = s2;
  if (c == 0) {
    float a = 0.f, b = 0.f;
    for (int i = 0; i < 32; ++i) { a += wcat[i] * bt[i]; b += wcat[32 + i] * bph[i]; }
    out[128] = a; out[129] = b;
  }
}

// Attention of _global_graph (p==0): f[n,m] = fth[n] + fph[m]; leaky_relu;
// softmax over m; + Ck.  One wave per (b,t).
__global__ __launch_bounds__(32) void att_kernel(const float* xcd1, const float* y,
                                                 const float* wtvbuf, const float* Ck,
                                                 float* att, int M2,
                                                 long long sYb, int sYc, int sYt) {
  int t = blockIdx.x, b = blockIdx.y;
  int lane = threadIdx.x;
  const float* wtv = wtvbuf;
  const float* wpv = wtvbuf + 64;
  int ln25 = lane < 25 ? lane : 0;
  int lnM  = lane < M2 ? lane : 0;
  float fth = wtvbuf[128];
  {
    const float* px = xcd1 + (long long)b * 409600 + (long long)t * 25 + ln25;
    float acc = 0.f;
    for (int c = 0; c < 64; ++c) acc += wtv[c] * px[c * 1600];
    fth += (lane < 25) ? acc : 0.f;
  }
  float fph = wtvbuf[129];
  {
    const float* py = y + (long long)b * sYb + (long long)t * sYt + lnM;
    float acc = 0.f;
    for (int c = 0; c < 64; ++c) acc += wpv[c] * py[c * sYc];
    fph += (lane < M2) ? acc : 0.f;
  }
  long long base = ((long long)b * 64 + t) * 25 * M2;
  for (int n = 0; n < 25; ++n) {
    float fn = __shfl(fth, n);
    float f = fn + fph;
    f = f >= 0.f ? f : 0.2f * f;
    float v = (lane < M2) ? f : -3.4e38f;
    float mx = v;
    for (int o = 16; o > 0; o >>= 1) mx = fmaxf(mx, __shfl_xor(mx, o));
    float e = (lane < M2) ? __expf(v - mx) : 0.f;
    float s = e;
    for (int o = 16; o > 0; o >>= 1) s += __shfl_xor(s, o);
    if (lane < M2) att[base + n * M2 + lane] = e / s + Ck[n * M2 + lane];
  }
}

// TCN 3x1 conv over T on the 4C concat tensor, fused bias + BN + residual + PReLU.
// One wave per 16x16 tile of the (64 out-ch x 25 joint) output for each (b,t).
// A = Wtcn viewed as (64, 768) row-major with k = i*3 + kk.
__global__ __launch_bounds__(32) void tcn_kernel(const float* __restrict__ W,
                                                 const float* __restrict__ Hcat,
                                                 const float* __restrict__ x,
                                                 const float* __restrict__ btcn,
                                                 const float* __restrict__ aoutP,
                                                 float* __restrict__ out) {
  int tile = blockIdx.x;                  // 8 = 4 m-tiles x 2 n-tiles
  int tm = (tile >> 1) << 4;
  int tn = (tile & 1) << 4;
  int t = blockIdx.y, b = blockIdx.z;
  int lane = threadIdx.x;
  int half = lane >> 4, lid = lane & 15;
  int am = tm + lid;                      // < 64 always
  int bn = tn + lid;
  int bnc = bn < 25 ? bn : 24;            // clamped column: loads legal; garbage
                                          // only reaches unstored D columns
  const float* Hb = Hcat + (long long)b * 409600 + bnc;
  v8f acc = {};
  for (int k0 = 0; k0 < 768; k0 += 32) {
    v16h af, bf;
#pragma unroll
    for (int j = 0; j < 16; ++j) {
      int kk = k0 + ((j >> 3) << 4) + (half << 3) + (j & 7);
      float a = W[am * 768 + kk];
      int i = kk / 3, r = kk - 3 * i;
      int ts = t + r - 1;
      bool ok = (unsigned)ts < 64u;       // T zero-padding only
      int tsc = ok ? ts : 0;
      float bv = Hb[(i * 64 + tsc) * 25];
      bv = ok ? bv : 0.f;
      af[j] = (_Float16)a;
      bf[j] = (_Float16)bv;
    }
    acc = __builtin_amdgcn_wmma_f32_16x16x32_f16(false, af, false, bf,
                                                 (short)0, acc, false, false);
  }
  float ap = *aoutP;
  if (bn < 25) {
#pragma unroll
    for (int r = 0; r < 8; ++r) {
      int cm = tm + r + (half << 3);
      long long oidx = ((long long)(b * 64 + cm) * 64 + t) * 25 + bn;
      float v = (acc[r] + btcn[cm]) * BN_SCALE + x[oidx];
      out[oidx] = v >= 0.f ? v : ap * v;
    }
  }
}

static GemmDesc mk(const float* A, const float* B, float* C, int M, int N, int K) {
  GemmDesc d{};
  d.A = A; d.B = B; d.C = C; d.M = M; d.N = N; d.K = K;
  d.alpha = 1.f; d.biasN = nullptr; d.preluA = nullptr; d.epilogue = 0;
  return d;
}
static void gemm(hipStream_t s, const GemmDesc& d, int b0, int b1) {
  dim3 g((unsigned)(((d.M + 15) >> 4) * ((d.N + 15) >> 4)), (unsigned)b1, (unsigned)b0);
  gemm_wmma_kernel<<<g, dim3(32), 0, s>>>(d);
}

extern "C" void kernel_launch(void* const* d_in, const int* in_sizes, int n_in,
                              void* d_out, int out_size, void* d_ws, size_t ws_size,
                              hipStream_t stream) {
  (void)in_sizes; (void)n_in; (void)out_size; (void)ws_size;
  const float* x     = (const float*)d_in[0];
  const float* A1    = (const float*)d_in[1];
  const float* A2    = (const float*)d_in[2];
  const float* A3    = (const float*)d_in[3];
  const float* Wp1   = (const float*)d_in[4];
  const float* ap1   = (const float*)d_in[5];
  const float* Wp2   = (const float*)d_in[6];
  const float* ap2   = (const float*)d_in[7];
  const float* Wg2   = (const float*)d_in[8];
  const float* bg2   = (const float*)d_in[9];
  const float* Wt2   = (const float*)d_in[10];
  const float* bt2   = (const float*)d_in[11];
  const float* Wph2  = (const float*)d_in[12];
  const float* bph2  = (const float*)d_in[13];
  const float* wcat2 = (const float*)d_in[14];
  const float* Ck2   = (const float*)d_in[15];
  const float* Wg3   = (const float*)d_in[16];
  const float* bg3   = (const float*)d_in[17];
  const float* Wt3   = (const float*)d_in[18];
  const float* bt3   = (const float*)d_in[19];
  const float* Wph3  = (const float*)d_in[20];
  const float* bph3  = (const float*)d_in[21];
  const float* wcat3 = (const float*)d_in[22];
  const float* Ck3   = (const float*)d_in[23];
  const float* Wtcn  = (const float*)d_in[24];
  const float* btcn  = (const float*)d_in[25];
  const float* aout  = (const float*)d_in[26];
  float* out = (float*)d_out;
  float* ws  = (float*)d_ws;

  // Workspace layout (floats). Hcat = concat(x, x_cd1, x_cd2r, x_cd3r) (B,256,64,25).
  float* Hcat = ws;                       // 26,214,400
  float* FA1  = ws + 26214400;            //  2,560,000  (B*T,25,25)
  float* Fp   = FA1 + 2560000;            //  6,553,600  (B,C,T,25)
  float* Gb   = Fp + 6553600;             //  6,553,600  permuted pool features
  float* Pb   = Gb + 6553600;             //     30,400  pool softmax weights
  float* X2   = Pb + 30400;               //  4,980,736  x_2 / x_3
  float* FA2  = X2 + 4980736;             //  1,478,656  FA2 / FA3
  float* Y2   = FA2 + 1478656;            //  4,980,736  x_cd2p / x_cd3p
  float* GX   = Y2 + 4980736;             //  4,980,736  g_x
  float* ATT  = GX + 4980736;             //  1,945,600  attention
  float* WTV  = ATT + 1945600;            //        130  reduced wcat weights

  float* S1 = Hcat + 102400;              // x_cd1 slice  (b-stride 409600)
  float* S2 = Hcat + 204800;              // x_cd2r slice
  float* S3 = Hcat + 307200;              // x_cd3r slice

  // (0) x -> Hcat slice 0
  copy_x_hcat<<<25600, 256, 0, stream>>>(x, Hcat);

  // ---- LDNET 1 ----
  { GemmDesc d = mk(x, x, FA1, 25, 25, 64);                 // gram
    d.sAb0=102400; d.sAb1=25; d.sAm=1;    d.sAk=1600;
    d.sBb0=102400; d.sBb1=25; d.sBk=1600; d.sBn=1;
    d.sCb0=40000;  d.sCb1=625; d.sCm=25;  d.sCn=1;
    d.alpha = 0.125f; gemm(stream, d, 64, 64); }
  softmax_rows<<<102400, 32, 0, stream>>>(FA1, 25, A1, 25);
  { GemmDesc d = mk(x, FA1, S1, 64, 25, 25);                // propagate -> x_cd1
    d.sAb0=102400; d.sAb1=25;  d.sAm=1600; d.sAk=1;
    d.sBb0=40000;  d.sBb1=625; d.sBk=25;   d.sBn=1;
    d.sCb0=409600; d.sCb1=25;  d.sCm=1600; d.sCn=1;
    gemm(stream, d, 64, 64); }

  // ---- Pool 1 (25 -> 19) ----
  { GemmDesc d = mk(S1, FA1, Fp, 64, 25, 25);               // F = prelu(Fn @ FA)
    d.sAb0=409600; d.sAb1=25;  d.sAm=1600; d.sAk=1;
    d.sBb0=40000;  d.sBb1=625; d.sBk=25;   d.sBn=1;
    d.sCb0=102400; d.sCb1=25;  d.sCm=1600; d.sCn=1;
    d.epilogue = 1; d.preluA = ap1; gemm(stream, d, 64, 64); }
  permute_pool<<<25600, 256, 0, stream>>>(Fp, Gb);
  { GemmDesc d = mk(Gb, Wp1, Pb, 25, 19, 4096);             // linear
    d.sAb0=102400; d.sAb1=0; d.sAm=1; d.sAk=25;
    d.sBb0=0; d.sBb1=0; d.sBk=1; d.sBn=4096;
    d.sCb0=475; d.sCb1=0; d.sCm=19; d.sCn=1;
    gemm(stream, d, 64, 1); }
  softmax_rows<<<1600, 32, 0, stream>>>(Pb, 19, nullptr, 1);
  { GemmDesc d = mk(S1, Pb, X2, 4096, 19, 25);              // x_2 = Fn @ F
    d.sAb0=409600; d.sAb1=0; d.sAm=25; d.sAk=1;
    d.sBb0=475;    d.sBb1=0; d.sBk=19; d.sBn=1;
    d.sCb0=77824;  d.sCb1=0; d.sCm=19; d.sCn=1;
    gemm(stream, d, 64, 1); }

  // ---- LDNET 2 ----
  { GemmDesc d = mk(X2, X2, FA2, 19, 19, 64);
    d.sAb0=77824; d.sAb1=19;  d.sAm=1;    d.sAk=1216;
    d.sBb0=77824; d.sBb1=19;  d.sBk=1216; d.sBn=1;
    d.sCb0=23104; d.sCb1=361; d.sCm=19;   d.sCn=1;
    d.alpha = 0.125f; gemm(stream, d, 64, 64); }
  softmax_rows<<<77824, 32, 0, stream>>>(FA2, 19, A2, 19);
  { GemmDesc d = mk(X2, FA2, Y2, 64, 19, 19);
    d.sAb0=77824; d.sAb1=19;  d.sAm=1216; d.sAk=1;
    d.sBb0=23104; d.sBb1=361; d.sBk=19;   d.sBn=1;
    d.sCb0=77824; d.sCb1=19;  d.sCm=1216; d.sCn=1;
    gemm(stream, d, 64, 64); }

  // ---- Restore 2 ----
  wtv_kernel<<<1, 64, 0, stream>>>(wcat2, Wt2, bt2, Wph2, bph2, WTV);
  { GemmDesc d = mk(Y2, Wg2, GX, 19, 64, 64);               // g_x
    d.sAb0=77824; d.sAb1=19;   d.sAm=1;  d.sAk=1216;
    d.sBb0=0;     d.sBb1=0;    d.sBk=1;  d.sBn=64;
    d.sCb0=77824; d.sCb1=1216; d.sCm=64; d.sCn=1;
    d.biasN = bg2; gemm(stream, d, 64, 64); }
  att_kernel<<<dim3(64, 64), 32, 0, stream>>>(S1, Y2, WTV, Ck2, ATT, 19, 77824LL, 1216, 19);
  { GemmDesc d = mk(GX, ATT, S2, 64, 25, 19);               // att @ g_x -> relu(BN)
    d.sAb0=77824;  d.sAb1=1216; d.sAm=1;    d.sAk=64;
    d.sBb0=30400;  d.sBb1=475;  d.sBk=1;    d.sBn=19;
    d.sCb0=409600; d.sCb1=25;   d.sCm=1600; d.sCn=1;
    d.epilogue = 2; gemm(stream, d, 64, 64); }

  // ---- Pool 2 (25 -> 15) ----
  { GemmDesc d = mk(S1, FA1, Fp, 64, 25, 25);
    d.sAb0=409600; d.sAb1=25;  d.sAm=1600; d.sAk=1;
    d.sBb0=40000;  d.sBb1=625; d.sBk=25;   d.sBn=1;
    d.sCb0=102400; d.sCb1=25;  d.sCm=1600; d.sCn=1;
    d.epilogue = 1; d.preluA = ap2; gemm(stream, d, 64, 64); }
  permute_pool<<<25600, 256, 0, stream>>>(Fp, Gb);
  { GemmDesc d = mk(Gb, Wp2, Pb, 25, 15, 4096);
    d.sAb0=102400; d.sAb1=0; d.sAm=1; d.sAk=25;
    d.sBb0=0; d.sBb1=0; d.sBk=1; d.sBn=4096;
    d.sCb0=375; d.sCb1=0; d.sCm=15; d.sCn=1;
    gemm(stream, d, 64, 1); }
  softmax_rows<<<1600, 32, 0, stream>>>(Pb, 15, nullptr, 1);
  { GemmDesc d = mk(S1, Pb, X2, 4096, 15, 25);              // x_3 (reuses X2)
    d.sAb0=409600; d.sAb1=0; d.sAm=25; d.sAk=1;
    d.sBb0=375;    d.sBb1=0; d.sBk=15; d.sBn=1;
    d.sCb0=61440;  d.sCb1=0; d.sCm=15; d.sCn=1;
    gemm(stream, d, 64, 1); }

  // ---- LDNET 3 ----
  { GemmDesc d = mk(X2, X2, FA2, 15, 15, 64);
    d.sAb0=61440; d.sAb1=15;  d.sAm=1;   d.sAk=960;
    d.sBb0=61440; d.sBb1=15;  d.sBk=960; d.sBn=1;
    d.sCb0=14400; d.sCb1=225; d.sCm=15;  d.sCn=1;
    d.alpha = 0.125f; gemm(stream, d, 64, 64); }
  softmax_rows<<<61440, 32, 0, stream>>>(FA2, 15, A3, 15);
  { GemmDesc d = mk(X2, FA2, Y2, 64, 15, 15);
    d.sAb0=61440; d.sAb1=15;  d.sAm=960; d.sAk=1;
    d.sBb0=14400; d.sBb1=225; d.sBk=15;  d.sBn=1;
    d.sCb0=61440; d.sCb1=15;  d.sCm=960; d.sCn=1;
    gemm(stream, d, 64, 64); }

  // ---- Restore 3 ----
  wtv_kernel<<<1, 64, 0, stream>>>(wcat3, Wt3, bt3, Wph3, bph3, WTV);
  { GemmDesc d = mk(Y2, Wg3, GX, 15, 64, 64);
    d.sAb0=61440; d.sAb1=15;  d.sAm=1;  d.sAk=960;
    d.sBb0=0;     d.sBb1=0;   d.sBk=1;  d.sBn=64;
    d.sCb0=61440; d.sCb1=960; d.sCm=64; d.sCn=1;
    d.biasN = bg3; gemm(stream, d, 64, 64); }
  att_kernel<<<dim3(64, 64), 32, 0, stream>>>(S1, Y2, WTV, Ck3, ATT, 15, 61440LL, 960, 15);
  { GemmDesc d = mk(GX, ATT, S3, 64, 25, 15);
    d.sAb0=61440;  d.sAb1=960; d.sAm=1;    d.sAk=64;
    d.sBb0=24000;  d.sBb1=375; d.sBk=1;    d.sBn=15;
    d.sCb0=409600; d.sCb1=25;  d.sCm=1600; d.sCn=1;
    d.epilogue = 2; gemm(stream, d, 64, 64); }

  // ---- TCN + BN + residual + PReLU -> d_out ----
  tcn_kernel<<<dim3(8, 64, 64), 32, 0, stream>>>(Wtcn, Hcat, x, btcn, aout, out);
}
